// Model_9320079032471
// MI455X (gfx1250) — matrix operand order repaired and checked by reading, compiled, not run on hardware
//
#include <hip/hip_runtime.h>
#include <hip/hip_bf16.h>

typedef __attribute__((ext_vector_type(16))) _Float16 v16h;
typedef __attribute__((ext_vector_type(8)))  _Float16 v8h;
typedef __attribute__((ext_vector_type(2)))  _Float16 h2;
typedef __attribute__((ext_vector_type(8)))  float    v8f;

#define NB    512          // batch
#define G0    63           // input grid
#define EPSL  1e-4f

struct BNP { const float *g, *b, *m, *v; };

// ---------------------------------------------------------------------------
// 1) scatter-add points into dense image + count
// ---------------------------------------------------------------------------
__global__ void k_scatter(const int* __restrict__ coords, const int* __restrict__ bidx,
                          const float* __restrict__ feat, float* __restrict__ x0,
                          float* __restrict__ cnt, int N) {
  int i = blockIdx.x * blockDim.x + threadIdx.x;
  if (i >= N) return;
  int r = coords[2 * i], c = coords[2 * i + 1], b = bidx[i];
  size_t idx = ((size_t)b * G0 + r) * G0 + c;
  atomicAdd(&x0[idx], feat[i]);
  atomicAdd(&cnt[idx], 1.0f);
}

// ---------------------------------------------------------------------------
// 2) fused conv0 (7x7, 1->16, SAME) * m  + masked 3x3/s2 maxpool -> 31x31x16
// ---------------------------------------------------------------------------
__global__ __launch_bounds__(256)
void k_conv0pool(const float* __restrict__ x0, const float* __restrict__ cnt,
                 const float* __restrict__ w, _Float16* __restrict__ x1,
                 float* __restrict__ m1) {
  __shared__ float sw[49 * 16];
  int tid = threadIdx.x;
  for (int i = tid; i < 49 * 16; i += 256) sw[i] = w[i];
  __syncthreads();
  int b  = blockIdx.y;
  int op = blockIdx.x * 256 + tid;
  if (op >= 31 * 31) return;
  int oy = op / 31, ox = op - oy * 31;
  const float* xb = x0 + (size_t)b * G0 * G0;
  const float* cb = cnt + (size_t)b * G0 * G0;
  float best[16];
#pragma unroll
  for (int c = 0; c < 16; ++c) best[c] = -1e30f;
  bool any = false;
  for (int wy = 0; wy < 3; ++wy)
    for (int wx = 0; wx < 3; ++wx) {
      int py = oy * 2 + wy, px = ox * 2 + wx;
      if (cb[py * G0 + px] <= 0.0f) continue;
      any = true;
      float acc[16];
#pragma unroll
      for (int c = 0; c < 16; ++c) acc[c] = 0.0f;
      for (int ky = 0; ky < 7; ++ky) {
        int iy = py + ky - 3;
        if ((unsigned)iy >= (unsigned)G0) continue;
        for (int kx = 0; kx < 7; ++kx) {
          int ix = px + kx - 3;
          if ((unsigned)ix >= (unsigned)G0) continue;
          float v = xb[iy * G0 + ix];
          const float* wk = &sw[(ky * 7 + kx) * 16];
#pragma unroll
          for (int c = 0; c < 16; ++c) acc[c] = fmaf(v, wk[c], acc[c]);
        }
      }
#pragma unroll
      for (int c = 0; c < 16; ++c) best[c] = fmaxf(best[c], acc[c]);
    }
  m1[(size_t)b * 961 + op] = any ? 1.0f : 0.0f;
  _Float16* o = x1 + ((size_t)b * 961 + op) * 16;
#pragma unroll
  for (int c = 0; c < 16; ++c) o[c] = (_Float16)(any ? best[c] : 0.0f);
}

// ---------------------------------------------------------------------------
// 3) 3x3 mask maxpool (VALID)
// ---------------------------------------------------------------------------
__global__ void k_maskpool(const float* __restrict__ mi, float* __restrict__ mo,
                           int Hi, int Wi, int Ho, int Wo, int stride) {
  int id = blockIdx.x * blockDim.x + threadIdx.x;
  int tot = NB * Ho * Wo;
  if (id >= tot) return;
  int b = id / (Ho * Wo), r = id - b * Ho * Wo;
  int oy = r / Wo, ox = r - oy * Wo;
  const float* m = mi + (size_t)b * Hi * Wi;
  float mx = 0.0f;
  for (int dy = 0; dy < 3; ++dy)
    for (int dx = 0; dx < 3; ++dx)
      mx = fmaxf(mx, m[(oy * stride + dy) * Wi + ox * stride + dx]);
  mo[id] = mx;
}

// ---------------------------------------------------------------------------
// 4) weight prep: HWIO f32 -> transposed [Cout][K padded to 32] f16, zero tail
// ---------------------------------------------------------------------------
__global__ void k_wprep(const float* __restrict__ src, _Float16* __restrict__ dst,
                        int Ktot, int KS, int Cout) {
  int i = blockIdx.x * blockDim.x + threadIdx.x;
  if (i >= Cout * KS) return;
  int co = i / KS, k = i - co * KS;
  dst[i] = (k < Ktot) ? (_Float16)src[(size_t)k * Cout + co] : (_Float16)0.0f;
}

// ---------------------------------------------------------------------------
// 5) templated fused 3x3 conv: implicit GEMM on v_wmma_f32_16x16x32_f16
//    out[p,co] = (sum_k A(p,k) * W(k,co)) * m_out[p] + res[p,co]
//    A staged in LDS with BN-ReLU-mask pre-applied (PREBN)
// ---------------------------------------------------------------------------
struct ConvArgsT {
  const _Float16 *in, *w, *res;
  _Float16 *out;
  const float *m_in, *m_out;
  const float *bn_g, *bn_b, *bn_m, *bn_v;
  int H, Wd, Ho, Wo;
};

template<int CIN, int COUT, int STRIDE, int PAD, int PREBN>
__global__ __launch_bounds__(128)
void k_conv_wmma(ConvArgsT A) {
  constexpr int KTOT = 9 * CIN;
  constexpr int KS   = (KTOT + 31) & ~31;       // K padded to 32 (B rows padded)
  __shared__ _Float16 s_img[31 * 31 * 16];      // largest input tile
  __shared__ float    s_scale[96];
  __shared__ float    s_bias[96];

  const int b   = blockIdx.x;
  const int tid = threadIdx.x;
  const int npix = A.H * A.Wd;

  if (PREBN) {
    for (int i = tid; i < CIN; i += 128) {
      float sc   = A.bn_g[i] * rsqrtf(A.bn_v[i] + EPSL);
      s_scale[i] = sc;
      s_bias[i]  = A.bn_b[i] - A.bn_m[i] * sc;
    }
    __syncthreads();
  }
  { // stage image into LDS, fusing the pre-activation when PREBN
    const h2* src = (const h2*)(A.in + (size_t)b * npix * CIN);
    h2*       dst = (h2*)s_img;
    const int np = npix * CIN / 2;
    if (PREBN) {
      const float* mb = A.m_in + (size_t)b * npix;
      for (int i = tid; i < np; i += 128) {
        h2 d = src[i];
        int e = 2 * i, pix = e / CIN, ci = e - pix * CIN;
        float mk = mb[pix];
        float f0 = (float)d.x * s_scale[ci]     + s_bias[ci];
        float f1 = (float)d.y * s_scale[ci + 1] + s_bias[ci + 1];
        h2 o;
        o.x = (_Float16)(mk > 0.0f ? fmaxf(f0, 0.0f) : 0.0f);
        o.y = (_Float16)(mk > 0.0f ? fmaxf(f1, 0.0f) : 0.0f);
        dst[i] = o;
      }
    } else {
      for (int i = tid; i < np; i += 128) dst[i] = src[i];
    }
  }
  __syncthreads();

  const int lane  = tid & 31, wave = tid >> 5;
  const int laneM = lane & 15;
  const int lg    = (tid >> 4) & 1;
  const int P = A.Ho * A.Wo;
  const int ptiles = (P + 15) >> 4, ctiles = COUT >> 4;

  for (int t = wave; t < ptiles * ctiles; t += 4) {
    const int pt = t / ctiles, ct = t - pt * ctiles;
    const int p  = pt * 16 + laneM;                  // A-row pixel for this lane
    const int oy = p / A.Wo, ox = p - (p / A.Wo) * A.Wo;
    const int iy0 = oy * STRIDE - PAD, ix0 = ox * STRIDE - PAD;
    const _Float16* wrow = A.w + (size_t)(ct * 16 + laneM) * KS;

    v8f acc = {0.f, 0.f, 0.f, 0.f, 0.f, 0.f, 0.f, 0.f};
#pragma unroll
    for (int s = 0; s < KS / 32; ++s) {
      const int kk = s * 32;
      // --- A fragment: two runs of 8 consecutive K (16B LDS loads) ---
      v8h a0 = {0, 0, 0, 0, 0, 0, 0, 0};
      v8h a1 = {0, 0, 0, 0, 0, 0, 0, 0};
      {
        int k0 = kk + lg * 8;
        if (k0 < KTOT) {
          int tap = k0 / CIN, ci = k0 - tap * CIN;
          int iy = iy0 + tap / 3, ix = ix0 + (tap - (tap / 3) * 3);
          if ((unsigned)iy < (unsigned)A.H && (unsigned)ix < (unsigned)A.Wd)
            a0 = *(const v8h*)(s_img + (iy * A.Wd + ix) * CIN + ci);
        }
      }
      {
        int k1 = kk + 16 + lg * 8;
        if (k1 < KTOT) {
          int tap = k1 / CIN, ci = k1 - tap * CIN;
          int iy = iy0 + tap / 3, ix = ix0 + (tap - (tap / 3) * 3);
          if ((unsigned)iy < (unsigned)A.H && (unsigned)ix < (unsigned)A.Wd)
            a1 = *(const v8h*)(s_img + (iy * A.Wd + ix) * CIN + ci);
        }
      }
      v16h av = __builtin_shufflevector(a0, a1, 0, 1, 2, 3, 4, 5, 6, 7,
                                                8, 9, 10, 11, 12, 13, 14, 15);
      // --- B fragment: two runs of 8 consecutive K from transposed weights (2x16B) ---
      v8h b0 = *(const v8h*)(wrow + kk + lg * 8);
      v8h b1 = *(const v8h*)(wrow + kk + 16 + lg * 8);
      v16h bv = __builtin_shufflevector(b0, b1, 0, 1, 2, 3, 4, 5, 6, 7,
                                                8, 9, 10, 11, 12, 13, 14, 15);
      acc = __builtin_amdgcn_wmma_f32_16x16x32_f16(false, av, false, bv,
                                                   (short)0, acc, false, false);
    }

    // epilogue: D row = r + 8*laneGroup, col = laneM
#pragma unroll
    for (int r = 0; r < 8; ++r) {
      int pp = pt * 16 + r + lg * 8;
      if (pp < P) {
        size_t o = ((size_t)b * P + pp) * COUT + ct * 16 + laneM;
        float f  = acc[r] * A.m_out[(size_t)b * P + pp];
        if (A.res) f += (float)A.res[o];
        A.out[o] = (_Float16)f;
      }
    }
  }
}

// ---------------------------------------------------------------------------
// 6) head: bn_final + relu + mask, 192x4 linear, softmax
// ---------------------------------------------------------------------------
__global__ void k_head(const _Float16* __restrict__ xf, const float* __restrict__ mf,
                       const float* __restrict__ g, const float* __restrict__ bb,
                       const float* __restrict__ mm, const float* __restrict__ vv,
                       const float* __restrict__ lw, const float* __restrict__ lb,
                       float* __restrict__ out) {
  int b = blockIdx.x * blockDim.x + threadIdx.x;
  if (b >= NB) return;
  const _Float16* x = xf + (size_t)b * 192;
  float mk = mf[b];
  float lg4[4] = {lb[0], lb[1], lb[2], lb[3]};
  for (int c = 0; c < 192; ++c) {
    float sc = g[c] * rsqrtf(vv[c] + EPSL);
    float y  = ((float)x[c] - mm[c]) * sc + bb[c];
    y = fmaxf(y, 0.0f) * mk;
#pragma unroll
    for (int j = 0; j < 4; ++j) lg4[j] = fmaf(y, lw[c * 4 + j], lg4[j]);
  }
  float mx = fmaxf(fmaxf(lg4[0], lg4[1]), fmaxf(lg4[2], lg4[3]));
  float e[4], s = 0.0f;
#pragma unroll
  for (int j = 0; j < 4; ++j) { e[j] = __expf(lg4[j] - mx); s += e[j]; }
#pragma unroll
  for (int j = 0; j < 4; ++j) out[b * 4 + j] = e[j] / s;
}

// ---------------------------------------------------------------------------
// host orchestration
// ---------------------------------------------------------------------------
extern "C" void kernel_launch(void* const* d_in, const int* in_sizes, int n_in,
                              void* d_out, int out_size, void* d_ws, size_t ws_size,
                              hipStream_t stream) {
  (void)n_in; (void)out_size; (void)ws_size;
  int ii = 0;
  const int*   coords = (const int*)d_in[ii++];
  const int*   bidx   = (const int*)d_in[ii++];
  const float* feats  = (const float*)d_in[ii++];
  const float* wconv0 = (const float*)d_in[ii++];

  auto getBN = [&]() {
    BNP r;
    r.g = (const float*)d_in[ii++]; r.b = (const float*)d_in[ii++];
    r.m = (const float*)d_in[ii++]; r.v = (const float*)d_in[ii++];
    return r;
  };

  const int cfg_in[4]  = {16, 16, 32, 64};
  const int cfg_out[4] = {16, 32, 64, 96};
  const int cfg_st[4]  = {1, 2, 2, 2};

  BNP bn_a[4][2], bn_b[4][2];
  const float *wc1[4][2], *wc2[4][2], *wcs[4] = {nullptr, nullptr, nullptr, nullptr};
  for (int li = 0; li < 4; ++li)
    for (int rep = 0; rep < 2; ++rep) {
      bn_a[li][rep] = getBN();
      wc1[li][rep]  = (const float*)d_in[ii++];
      bn_b[li][rep] = getBN();
      wc2[li][rep]  = (const float*)d_in[ii++];
      if (rep == 0 && cfg_st[li] > 1) wcs[li] = (const float*)d_in[ii++];
    }
  BNP bn_res = getBN();
  const float* wfin = (const float*)d_in[ii++];
  BNP bn_fin = getBN();
  const float* lin_w = (const float*)d_in[ii++];
  const float* lin_b = (const float*)d_in[ii++];

  // ---- workspace bump allocator ----
  char* wsb = (char*)d_ws; size_t off = 0;
  auto alloc = [&](size_t n) { void* p = wsb + off; off = (off + n + 255) & ~(size_t)255; return p; };

  float* x0  = (float*)alloc((size_t)NB * G0 * G0 * 4);
  float* cnt = (float*)alloc((size_t)NB * G0 * G0 * 4);
  float* m31 = (float*)alloc((size_t)NB * 961 * 4);
  float* m15 = (float*)alloc((size_t)NB * 225 * 4);
  float* m7  = (float*)alloc((size_t)NB * 49 * 4);
  float* m3  = (float*)alloc((size_t)NB * 9 * 4);
  float* mF  = (float*)alloc((size_t)NB * 4);
  const size_t ACT = (size_t)NB * 961 * 16 * 2;
  _Float16* bufA = (_Float16*)alloc(ACT);
  _Float16* bufU = (_Float16*)alloc(ACT);
  _Float16* bufY = (_Float16*)alloc(ACT);
  _Float16* bufS = (_Float16*)alloc(ACT);

  // transpose+pad+cast weights: [Cout][KS] f16
  auto prep = [&](const float* src, int Cin, int Cout) {
    int Ktot = 9 * Cin, KS = (Ktot + 31) & ~31;
    int n = Cout * KS;
    _Float16* d = (_Float16*)alloc((size_t)n * 2);
    k_wprep<<<(n + 255) / 256, 256, 0, stream>>>(src, d, Ktot, KS, Cout);
    return d;
  };
  _Float16 *hc1[4][2], *hc2[4][2], *hcs[4] = {nullptr, nullptr, nullptr, nullptr};
  for (int li = 0; li < 4; ++li)
    for (int rep = 0; rep < 2; ++rep) {
      int cin = rep ? cfg_out[li] : cfg_in[li], cout = cfg_out[li];
      hc1[li][rep] = prep(wc1[li][rep], cin, cout);
      hc2[li][rep] = prep(wc2[li][rep], cout, cout);
      if (rep == 0 && cfg_st[li] > 1) hcs[li] = prep(wcs[li], cin, cout);
    }
  _Float16* hfin = prep(wfin, 96, 192);

  // ---- pipeline ----
  hipMemsetAsync(x0, 0, (size_t)NB * G0 * G0 * 4, stream);
  hipMemsetAsync(cnt, 0, (size_t)NB * G0 * G0 * 4, stream);
  int N = in_sizes[0] / 2;
  k_scatter<<<(N + 255) / 256, 256, 0, stream>>>(coords, bidx, feats, x0, cnt, N);
  k_conv0pool<<<dim3(4, NB), 256, 0, stream>>>(x0, cnt, wconv0, bufA, m31);

  _Float16* x = bufA; _Float16* y = bufY;
  float* mcur = m31;

#define LAUNCH_CONV(CI, CO, ST, PD, PB, pin, pout, pw, pres, pmi, pmo, pbn, HH, HO) \
  do { ConvArgsT a_; a_.in = (pin); a_.out = (pout); a_.w = (pw); a_.res = (pres);  \
       a_.m_in = (pmi); a_.m_out = (pmo);                                           \
       const BNP* bp_ = (pbn);                                                      \
       if (bp_) { a_.bn_g = bp_->g; a_.bn_b = bp_->b; a_.bn_m = bp_->m; a_.bn_v = bp_->v; } \
       else     { a_.bn_g = a_.bn_b = a_.bn_m = a_.bn_v = nullptr; }                \
       a_.H = (HH); a_.Wd = (HH); a_.Ho = (HO); a_.Wo = (HO);                       \
       k_conv_wmma<CI, CO, ST, PD, PB><<<NB, 128, 0, stream>>>(a_); } while (0)

#define SWAPXY() do { _Float16* t_ = x; x = y; y = t_; } while (0)

  // ---- stage 0: 31x31x16, two identity reps ----
  LAUNCH_CONV(16, 16, 1, 1, 1, x, bufU, hc1[0][0], nullptr, mcur, mcur, &bn_a[0][0], 31, 31);
  LAUNCH_CONV(16, 16, 1, 1, 1, bufU, y, hc2[0][0], x,       mcur, mcur, &bn_b[0][0], 31, 31);
  SWAPXY();
  LAUNCH_CONV(16, 16, 1, 1, 1, x, bufU, hc1[0][1], nullptr, mcur, mcur, &bn_a[0][1], 31, 31);
  LAUNCH_CONV(16, 16, 1, 1, 1, bufU, y, hc2[0][1], x,       mcur, mcur, &bn_b[0][1], 31, 31);
  SWAPXY();

  // ---- stage 1: 31 -> 15, 16 -> 32 ----
  k_maskpool<<<(NB * 225 + 255) / 256, 256, 0, stream>>>(mcur, m15, 31, 31, 15, 15, 2);
  LAUNCH_CONV(16, 32, 2, 0, 0, x, bufS, hcs[1],    nullptr, mcur, m15, (const BNP*)nullptr, 31, 15);
  LAUNCH_CONV(16, 32, 2, 0, 1, x, bufU, hc1[1][0], nullptr, mcur, m15, &bn_a[1][0],         31, 15);
  LAUNCH_CONV(32, 32, 1, 1, 1, bufU, y, hc2[1][0], bufS,    m15,  m15, &bn_b[1][0],         15, 15);
  SWAPXY(); mcur = m15;
  LAUNCH_CONV(32, 32, 1, 1, 1, x, bufU, hc1[1][1], nullptr, mcur, mcur, &bn_a[1][1], 15, 15);
  LAUNCH_CONV(32, 32, 1, 1, 1, bufU, y, hc2[1][1], x,       mcur, mcur, &bn_b[1][1], 15, 15);
  SWAPXY();

  // ---- stage 2: 15 -> 7, 32 -> 64 ----
  k_maskpool<<<(NB * 49 + 255) / 256, 256, 0, stream>>>(mcur, m7, 15, 15, 7, 7, 2);
  LAUNCH_CONV(32, 64, 2, 0, 0, x, bufS, hcs[2],    nullptr, mcur, m7, (const BNP*)nullptr, 15, 7);
  LAUNCH_CONV(32, 64, 2, 0, 1, x, bufU, hc1[2][0], nullptr, mcur, m7, &bn_a[2][0],         15, 7);
  LAUNCH_CONV(64, 64, 1, 1, 1, bufU, y, hc2[2][0], bufS,    m7,   m7, &bn_b[2][0],         7,  7);
  SWAPXY(); mcur = m7;
  LAUNCH_CONV(64, 64, 1, 1, 1, x, bufU, hc1[2][1], nullptr, mcur, mcur, &bn_a[2][1], 7, 7);
  LAUNCH_CONV(64, 64, 1, 1, 1, bufU, y, hc2[2][1], x,       mcur, mcur, &bn_b[2][1], 7, 7);
  SWAPXY();

  // ---- stage 3: 7 -> 3, 64 -> 96 ----
  k_maskpool<<<(NB * 9 + 255) / 256, 256, 0, stream>>>(mcur, m3, 7, 7, 3, 3, 2);
  LAUNCH_CONV(64, 96, 2, 0, 0, x, bufS, hcs[3],    nullptr, mcur, m3, (const BNP*)nullptr, 7, 3);
  LAUNCH_CONV(64, 96, 2, 0, 1, x, bufU, hc1[3][0], nullptr, mcur, m3, &bn_a[3][0],         7, 3);
  LAUNCH_CONV(96, 96, 1, 1, 1, bufU, y, hc2[3][0], bufS,    m3,   m3, &bn_b[3][0],         3, 3);
  SWAPXY(); mcur = m3;
  LAUNCH_CONV(96, 96, 1, 1, 1, x, bufU, hc1[3][1], nullptr, mcur, mcur, &bn_a[3][1], 3, 3);
  LAUNCH_CONV(96, 96, 1, 1, 1, bufU, y, hc2[3][1], x,       mcur, mcur, &bn_b[3][1], 3, 3);
  SWAPXY();

  // ---- final: bn_res -> conv_final (3x3 VALID, 96->192) * mF -> head ----
  k_maskpool<<<2, 256, 0, stream>>>(mcur, mF, 3, 3, 1, 1, 1);
  LAUNCH_CONV(96, 192, 1, 0, 1, x, bufU, hfin, nullptr, mcur, mF, &bn_res, 3, 1);
  k_head<<<2, 256, 0, stream>>>(bufU, mF, bn_fin.g, bn_fin.b, bn_fin.m, bn_fin.v,
                                lin_w, lin_b, (float*)d_out);

#undef LAUNCH_CONV
#undef SWAPXY
}
